// PolarProjectionDepth_23725399343697
// MI455X (gfx1250) — matrix operand order repaired and checked by reading, not compiled, not run on hardware
//
#include <hip/hip_runtime.h>
#include <stdint.h>

// Shapes fixed by the reference harness.
#define B_ 32
#define D_ 64
#define H_ 64
#define W_ 128
#define S_ 64
#define Z_ 64

typedef __attribute__((ext_vector_type(16))) _Float16 v16h;
typedef __attribute__((ext_vector_type(8)))  float    v8f;

union FragU { uint32_t u[8]; v16h h; };          // 8 dwords <-> 16 halves
union PairU { _Float16 f[2]; uint32_t u; };      // packed f16 pair

// One workgroup per (b, w): 4096 blocks, 256 threads (8 wave32s).
// Phase 1: stage polar (h,s) f32 + image (d,h) f16-packed tiles in LDS.
// Phase 2: depth interpolation -> scores (h,z).
// Phase 3: softmax over h per column z (+ cell_score = logsumexp), probs
//          written as packed (h even|h odd) f16 pairs -> B-fragment-ready.
// Phase 4: C(64x64) = A(64x64,f16) x P(64x64,f16) via v_wmma_f32_16x16x32_f16.
//          A frags via 2x ds_load_b128, B frags via 8x ds_load_b32; each wave
//          reuses its A fragment across its two z-tiles.
__global__ __launch_bounds__(256)
void PolarProjectionDepth_kernel(const float* __restrict__ image,   // (B,D,H,W)
                                 const float* __restrict__ polar,   // (B,H,W,S)
                                 float* __restrict__ out_img,       // (B,D,Z,W)
                                 float* __restrict__ out_cell)      // (B,W,Z)
{
    __shared__ __align__(16) float    s_raw[H_][S_ + 4];        // raw polar (h,s)
    __shared__ float    s_sc[H_][Z_ + 1];                       // scores, then exp()
    __shared__ uint32_t s_pw[H_ / 2][Z_ + 4];                   // probs: f16 pair (h, h+1) @ col z
    __shared__ __align__(16) uint32_t s_aw[D_][(H_ + 8) / 2];   // image: f16 pair (h, h+1), 36 w/row

    const int t = threadIdx.x;
    const int b = blockIdx.x >> 7;         // / W_
    const int w = blockIdx.x & (W_ - 1);

    // ---- Phase 1: global -> LDS staging -------------------------------
    {
        const int h  = t >> 2;
        const int s0 = (t & 3) * 16;
        const float* src = polar + (((size_t)(b * H_ + h) * W_ + w) * S_ + s0);
        #pragma unroll
        for (int i = 0; i < 16; i += 4) {              // b128 global load + b128 LDS store
            *(float4*)&s_raw[h][s0 + i] = *(const float4*)(src + i);
        }
        const int d  = t >> 2;
        const int h0 = (t & 3) * 16;
        const float* isrc = image + ((size_t)(b * D_ + d) * H_ + h0) * W_ + w;
        #pragma unroll
        for (int i = 0; i < 16; i += 2) {              // strided reads; packed b32 LDS store
            PairU p;
            p.f[0] = (_Float16)isrc[(size_t)(i + 0) * W_];
            p.f[1] = (_Float16)isrc[(size_t)(i + 1) * W_];
            s_aw[d][(h0 + i) >> 1] = p.u;
        }
    }
    __syncthreads();

    // ---- Phase 2: fixed-position linear interpolation over S ----------
    {
        const int h  = t >> 2;
        const int z0 = (t & 3) * 16;
        #pragma unroll
        for (int i = 0; i < 16; ++i) {
            const int z = z0 + i;
            float pos = 10.5f * __log2f((float)(z + 1));   // (S-1)*log2(step/zmin)/6
            int i0 = (int)floorf(pos);
            i0 = i0 < 0 ? 0 : (i0 > S_ - 1 ? S_ - 1 : i0);
            const int i1 = (i0 + 1 > S_ - 1) ? S_ - 1 : i0 + 1;
            const float f = pos - (float)i0;
            s_sc[h][z] = s_raw[h][i0] * (1.0f - f) + s_raw[h][i1] * f;
        }
    }
    __syncthreads();

    // ---- Phase 3: softmax / logsumexp over h (thread t owns column z=t)
    if (t < Z_) {
        const int z = t;
        float m = -3.402823466e38f;
        for (int h = 0; h < H_; ++h) m = fmaxf(m, s_sc[h][z]);
        float sum = 0.0f;
        for (int h = 0; h < H_; ++h) {
            const float e = __expf(s_sc[h][z] - m);
            s_sc[h][z] = e;                            // same-thread in-place: safe
            sum += e;
        }
        const float inv = 1.0f / sum;
        for (int h = 0; h < H_; h += 2) {              // B-fragment-ready packed pairs
            PairU p;
            p.f[0] = (_Float16)(s_sc[h + 0][z] * inv);
            p.f[1] = (_Float16)(s_sc[h + 1][z] * inv);
            s_pw[h >> 1][z] = p.u;
        }
        out_cell[((size_t)b * W_ + w) * Z_ + z] = m + __logf(sum);
    }
    __syncthreads();

    // ---- Phase 4: WMMA GEMM, C(D,Z) = A(D,H) x P(H,Z), f32 accumulate -
    const int lane = t & 31;
    const int wv   = t >> 5;
    const int m16  = lane & 15;
    const int hi16 = lane >> 4;                 // 0 for lanes 0-15, 1 for 16-31
    const int d0   = (wv >> 1) * 16;            // a wave's two tiles share d0
    const int z0a  = ((wv & 1) * 2 + 0) * 16;
    const int z0b  = ((wv & 1) * 2 + 1) * 16;

    v8f c0 = {0.f, 0.f, 0.f, 0.f, 0.f, 0.f, 0.f, 0.f};
    v8f c1 = {0.f, 0.f, 0.f, 0.f, 0.f, 0.f, 0.f, 0.f};

    #pragma unroll
    for (int kc = 0; kc < H_; kc += 32) {       // K = 64 in two 16x16x32 steps
        // A fragment: lane needs K runs {hi*8..hi*8+7} and {16+hi*8..+7}
        // -> two 16B-aligned uint4 LDS loads (ds_load_b128).
        FragU af;
        const uint4 ra0 = *(const uint4*)&s_aw[d0 + m16][(kc + hi16 * 8) >> 1];
        const uint4 ra1 = *(const uint4*)&s_aw[d0 + m16][(kc + 16 + hi16 * 8) >> 1];
        af.u[0] = ra0.x; af.u[1] = ra0.y; af.u[2] = ra0.z; af.u[3] = ra0.w;
        af.u[4] = ra1.x; af.u[5] = ra1.y; af.u[6] = ra1.z; af.u[7] = ra1.w;

        // B fragments: VGPR v holds K = {2v, 2v+1} (+8 for hi lanes, +8 more for v>=4) at col z
        // -> 8 ds_load_b32 each, already packed.
        FragU bf0, bf1;
        const int r0 = (kc >> 1) + hi16 * 4;
        #pragma unroll
        for (int v = 0; v < 8; ++v) {
            bf0.u[v] = s_pw[r0 + v + ((v & 4) ? 4 : 0)][z0a + m16];
            bf1.u[v] = s_pw[r0 + v + ((v & 4) ? 4 : 0)][z0b + m16];
        }
        c0 = __builtin_amdgcn_wmma_f32_16x16x32_f16(false, af.h, false, bf0.h,
                                                    (short)0, c0, false, false);
        c1 = __builtin_amdgcn_wmma_f32_16x16x32_f16(false, af.h, false, bf1.h,
                                                    (short)0, c1, false, false);
    }

    // C/D layout: VGPR r -> M = r + 8*(lane>=16); N = lane&15
    #pragma unroll
    for (int r = 0; r < 8; ++r) {
        const int d = d0 + (hi16 << 3) + r;
        out_img[((size_t)(b * D_ + d) * Z_ + (z0a + m16)) * W_ + w] = c0[r];
        out_img[((size_t)(b * D_ + d) * Z_ + (z0b + m16)) * W_ + w] = c1[r];
    }
}

extern "C" void kernel_launch(void* const* d_in, const int* in_sizes, int n_in,
                              void* d_out, int out_size, void* d_ws, size_t ws_size,
                              hipStream_t stream) {
    const float* image = (const float*)d_in[0];   // (32,64,64,128) f32
    const float* polar = (const float*)d_in[1];   // (32,64,128,64) f32
    float* out      = (float*)d_out;
    float* out_img  = out;                                   // 32*64*64*128 floats
    float* out_cell = out + (size_t)B_ * D_ * Z_ * W_;       // then 32*128*64 floats

    dim3 grid(B_ * W_);   // 4096 blocks: one per (b, w)
    dim3 block(256);      // 8 wave32s
    PolarProjectionDepth_kernel<<<grid, block, 0, stream>>>(image, polar, out_img, out_cell);
}
